// FBPINN_ansatz_59279138619964
// MI455X (gfx1250) — matrix operand order repaired and checked by reading, not compiled, not run on hardware
//
#include <hip/hip_runtime.h>

#define NPTS   131072
#define SSUB   32
#define HDIM   32
#define BLOCK  256
#define WAVES  8

typedef __attribute__((ext_vector_type(16))) _Float16 v16h;
typedef __attribute__((ext_vector_type(8)))  _Float16 v8h;
typedef __attribute__((ext_vector_type(2)))  _Float16 h2;
typedef __attribute__((ext_vector_type(8)))  float    v8f;

union AOp { v16h v; v8h h[2]; };
union BOp { v16h v; unsigned int u[8]; };

__device__ __forceinline__ float fast_rcp(float x) { return __builtin_amdgcn_rcpf(x); }

__device__ __forceinline__ float fast_tanh(float x) {
#if __has_builtin(__builtin_amdgcn_tanhf)
  return __builtin_amdgcn_tanhf(x);           // gfx1250 v_tanh_f32
#else
  return fmaf(-2.0f, fast_rcp(__expf(2.0f * x) + 1.0f), 1.0f);
#endif
}

__device__ __forceinline__ float fast_sigmoid(float x) {
  return fast_rcp(1.0f + __expf(-x));
}

// v_cvt_pk_rtz_f16_f32: pack two f32 into one VGPR of two halves.
// Builtin returns an __fp16 vector; bit-cast to our _Float16 vector type.
__device__ __forceinline__ h2 pk_f16(float a, float b) {
  return __builtin_bit_cast(h2, __builtin_amdgcn_cvt_pkrtz(a, b));
}

// Load both 16x32 f16 A operands of this wave's 32x32 tile (row-major LDS).
__device__ __forceinline__ void load_a(const _Float16 (*hrow)[HDIM], int lane,
                                       AOp& a0, AOp& a1) {
  const int l15  = lane & 15;
  const int lsel = (lane < 16) ? 0 : 16;  // K half-select (in halves)
  a0.h[0] = *(const v8h*)&hrow[l15][lsel];
  a0.h[1] = *(const v8h*)&hrow[l15][8 + lsel];
  a1.h[0] = *(const v8h*)&hrow[16 + l15][lsel];
  a1.h[1] = *(const v8h*)&hrow[16 + l15][8 + lsel];
}

// One hidden layer for a 32-point tile: h <- tanh(h @ W + b).
__device__ __forceinline__ void mlp_layer(_Float16 (*hrow)[HDIM], int lane,
                                          v16h bn0, v16h bn1,
                                          const float* bias) {
  const int l15  = lane & 15;
  const int lsel = (lane < 16) ? 0 : 8;

  AOp a0, a1;
  load_a(hrow, lane, a0, a1);

  // C = bias broadcast: C-layout N = lane%16 for every VGPR.
  const float bv0 = bias[l15];
  const float bv1 = bias[16 + l15];
  v8f c00, c01, c10, c11;
#pragma unroll
  for (int r = 0; r < 8; ++r) { c00[r] = bv0; c01[r] = bv1; c10[r] = bv0; c11[r] = bv1; }

  c00 = __builtin_amdgcn_wmma_f32_16x16x32_f16(false, a0.v, false, bn0, (short)0, c00, false, false);
  c01 = __builtin_amdgcn_wmma_f32_16x16x32_f16(false, a0.v, false, bn1, (short)0, c01, false, false);
  c10 = __builtin_amdgcn_wmma_f32_16x16x32_f16(false, a1.v, false, bn0, (short)0, c10, false, false);
  c11 = __builtin_amdgcn_wmma_f32_16x16x32_f16(false, a1.v, false, bn1, (short)0, c11, false, false);

  __builtin_amdgcn_wave_barrier();   // keep A loads ahead of the stores below

  // tanh + packed f16 convert + write back (C layout: M = r (+8 for hi lanes)).
  const int mb = lsel;               // 0 or 8
#pragma unroll
  for (int r = 0; r < 8; ++r) {
    h2 p0 = pk_f16(fast_tanh(c00[r]), fast_tanh(c01[r]));
    h2 p1 = pk_f16(fast_tanh(c10[r]), fast_tanh(c11[r]));
    hrow[mb + r][l15]           = p0[0];
    hrow[mb + r][16 + l15]      = p0[1];
    hrow[16 + mb + r][l15]      = p1[0];
    hrow[16 + mb + r][16 + l15] = p1[1];
  }
}

__global__ __launch_bounds__(BLOCK) void fbpinn_ansatz_kernel(
    const float* __restrict__ X,        const float* __restrict__ centers,
    const float* __restrict__ widths,   const float* __restrict__ core_min,
    const float* __restrict__ core_max, const float* __restrict__ overlap,
    const float* __restrict__ W1,  const float* __restrict__ b1,
    const float* __restrict__ W2,  const float* __restrict__ b2,
    const float* __restrict__ W3,  const float* __restrict__ b3,
    const float* __restrict__ Wout, const float* __restrict__ bout,
    const float* __restrict__ scale, const float* __restrict__ shift,
    float* __restrict__ out)
{
  __shared__ __align__(16) _Float16 hbuf[WAVES][32][HDIM]; // per-wave 32x32 f16 tiles
  __shared__ __align__(16) _Float16 w2T[HDIM][HDIM];       // [n][k] transposed f16
  __shared__ __align__(16) _Float16 w3T[HDIM][HDIM];
  __shared__ __align__(16) float ybuf[WAVES][32];          // per-wave head outputs
  __shared__ float w1s[2 * HDIM];
  __shared__ float b1s[HDIM], b2s[HDIM], b3s[HDIM], wouts[HDIM];

  const int tid  = threadIdx.x;
  const int lane = tid & 31;
  const int wave = tid >> 5;
  const int n    = blockIdx.x * BLOCK + tid;

  const float t = X[2 * n + 0];
  const float x = X[2 * n + 1];
  const float scl = scale[0], shf = shift[0];

  _Float16 (*hrow)[HDIM] = hbuf[wave];
  const int l15 = lane & 15;

  float acc = 0.0f, wsum = 0.0f;

  for (int s = 0; s < SSUB; ++s) {
    __syncthreads();  // previous iteration done reading staged weights
    // --- cooperative stage of subdomain-s weights (transpose + f16 convert) ---
#pragma unroll
    for (int i = 0; i < 4; ++i) {
      int idx = tid + i * BLOCK;            // idx = k*32 + c over W[s][k][c]
      int k = idx >> 5, c = idx & 31;
      w2T[c][k] = (_Float16)W2[s * HDIM * HDIM + idx];
      w3T[c][k] = (_Float16)W3[s * HDIM * HDIM + idx];
    }
    if (tid < 2 * HDIM) w1s[tid] = W1[s * 2 * HDIM + tid];
    if (tid < HDIM) {
      b1s[tid]   = b1[s * HDIM + tid];
      b2s[tid]   = b2[s * HDIM + tid];
      b3s[tid]   = b3[s * HDIM + tid];
      wouts[tid] = Wout[s * HDIM + tid];
    }
    __syncthreads();

    // --- window weight w_s(point) ---
    float ww = 1.0f;
#pragma unroll
    for (int d = 0; d < 2; ++d) {
      float Xd = (d == 0) ? t : x;
      float sf = 4.0f * fast_rcp(2.0f * overlap[s * 2 + d] * widths[s * 2 + d] + 1e-8f);
      float lg = fast_sigmoid(sf * (Xd - core_min[s * 2 + d]));
      float rg = fast_sigmoid(sf * (core_max[s * 2 + d] - Xd));
      ww *= lg * rg;
    }

    // --- layer 1 (K=2, VALU per-lane; lane == point within wave tile) ---
    float xn0 = (t - centers[s * 2 + 0]) * fast_rcp(widths[s * 2 + 0]) + 0.5f;
    float xn1 = (x - centers[s * 2 + 1]) * fast_rcp(widths[s * 2 + 1]) + 0.5f;
#pragma unroll
    for (int j = 0; j < HDIM; j += 2) {
      float ha = fast_tanh(fmaf(xn0, w1s[j],     fmaf(xn1, w1s[HDIM + j],     b1s[j])));
      float hb = fast_tanh(fmaf(xn0, w1s[j + 1], fmaf(xn1, w1s[HDIM + j + 1], b1s[j + 1])));
      *(h2*)&hrow[lane][j] = pk_f16(ha, hb);
    }

    // --- B operands for W2/W3 from transposed LDS (packed K-pairs, b32 loads) ---
    BOp b2n0, b2n1, b3n0, b3n1;
    {
      const int kb = (lane < 16) ? 0 : 16;
      const int n0 = l15, n1 = 16 + l15;
#pragma unroll
      for (int v = 0; v < 8; ++v) {
        b2n0.u[v] = *(const unsigned int*)&w2T[n0][kb + 2 * v];
        b2n1.u[v] = *(const unsigned int*)&w2T[n1][kb + 2 * v];
        b3n0.u[v] = *(const unsigned int*)&w3T[n0][kb + 2 * v];
        b3n1.u[v] = *(const unsigned int*)&w3T[n1][kb + 2 * v];
      }
    }

    // --- hidden layers 2 & 3 on WMMA ---
    mlp_layer(hrow, lane, b2n0.v, b2n1.v, b2s);
    mlp_layer(hrow, lane, b3n0.v, b3n1.v, b3s);

    // --- output head on WMMA: B has Wout in column 0, zeros elsewhere ---
    BOp bo;
    {
      const int kb = (lane < 16) ? 0 : 16;
      const unsigned keep = (l15 == 0) ? 0xFFFFFFFFu : 0u;
#pragma unroll
      for (int v = 0; v < 8; ++v) {
        h2 p = pk_f16(wouts[kb + 2 * v], wouts[kb + 2 * v + 1]);
        bo.u[v] = __builtin_bit_cast(unsigned int, p) & keep;
      }
    }
    AOp a0, a1;
    load_a(hrow, lane, a0, a1);
    v8f y0, y1;
#pragma unroll
    for (int r = 0; r < 8; ++r) { y0[r] = 0.0f; y1[r] = 0.0f; }
    y0 = __builtin_amdgcn_wmma_f32_16x16x32_f16(false, a0.v, false, bo.v, (short)0, y0, false, false);
    y1 = __builtin_amdgcn_wmma_f32_16x16x32_f16(false, a1.v, false, bo.v, (short)0, y1, false, false);

    // y for point M lives in column 0 => lanes 0 (M=r) and 16 (M=8+r); scatter
    // back to the lane-per-point mapping through LDS.
    if (l15 == 0) {
      const int hi = (lane < 16) ? 0 : 8;
#pragma unroll
      for (int r = 0; r < 8; ++r) {
        ybuf[wave][hi + r]      = y0[r];
        ybuf[wave][16 + hi + r] = y1[r];
      }
    }
    __builtin_amdgcn_wave_barrier();   // same-wave DS ops stay in order
    float dot = ybuf[wave][lane];

    float y = fmaf(scl, dot + bout[s], shf);
    acc  = fmaf(ww, y, acc);
    wsum += ww;
  }

  // --- normalize + ansatz ---
  float u = acc * fast_rcp(wsum + 1e-8f);
  float factor = fast_tanh(x + 1.0f) * fast_tanh(x - 1.0f) * fast_tanh(t);
  out[n] = fmaf(factor, u, -__sinf(3.14159265358979f * x));
}

extern "C" void kernel_launch(void* const* d_in, const int* in_sizes, int n_in,
                              void* d_out, int out_size, void* d_ws, size_t ws_size,
                              hipStream_t stream) {
  (void)in_sizes; (void)n_in; (void)d_ws; (void)ws_size; (void)out_size;
  const float* X        = (const float*)d_in[0];
  const float* centers  = (const float*)d_in[1];
  const float* widths   = (const float*)d_in[2];
  const float* core_min = (const float*)d_in[3];
  const float* core_max = (const float*)d_in[4];
  const float* overlap  = (const float*)d_in[5];
  const float* W1       = (const float*)d_in[6];
  const float* b1       = (const float*)d_in[7];
  const float* W2       = (const float*)d_in[8];
  const float* b2       = (const float*)d_in[9];
  const float* W3       = (const float*)d_in[10];
  const float* b3       = (const float*)d_in[11];
  const float* Wout     = (const float*)d_in[12];
  const float* bout     = (const float*)d_in[13];
  const float* scale    = (const float*)d_in[14];
  const float* shift    = (const float*)d_in[15];

  dim3 grid(NPTS / BLOCK);
  dim3 block(BLOCK);
  fbpinn_ansatz_kernel<<<grid, block, 0, stream>>>(
      X, centers, widths, core_min, core_max, overlap,
      W1, b1, W2, b2, W3, b3, Wout, bout, scale, shift,
      (float*)d_out);
}
